// BPA_4750233830209
// MI455X (gfx1250) — matrix operand order repaired and checked by reading, not compiled, not run on hardware
//
#include <hip/hip_runtime.h>
#include <math.h>

typedef __attribute__((ext_vector_type(16))) __bf16 v16bf;
typedef __attribute__((ext_vector_type(8)))  float  v8f;

// ---------- helpers ----------
__device__ __forceinline__ unsigned short f2bf(float f) {
    unsigned int u = __float_as_uint(f);
    u += 0x7FFFu + ((u >> 16) & 1u);   // round-to-nearest-even
    return (unsigned short)(u >> 16);
}
__device__ __forceinline__ float sigmoidf(float v) { return 1.0f / (1.0f + __expf(-v)); }

#define C_CH   1024
#define HW     1024
#define K2     2048

// ---------- 1) gaussian table: gus[p, s] (bf16, row-normalized) ----------
__global__ void gauss_kernel(unsigned short* __restrict__ gus) {
    __shared__ float sred[256];
    const int p = blockIdx.x;
    const int pi = p >> 5, pk = p & 31;
    const int t = threadIdx.x;
    float g[4]; float lsum = 0.f;
    #pragma unroll
    for (int q = 0; q < 4; ++q) {
        int s = t + q * 256;
        int sx = s >> 5, sy = s & 31;
        float d2 = (float)((pi - sx) * (pi - sx) + (pk - sy) * (pk - sy));
        g[q] = __expf(-d2 / 4.5f);          // 2*sigma^2 = 4.5 ; constant factor cancels
        lsum += g[q];
    }
    sred[t] = lsum; __syncthreads();
    for (int off = 128; off > 0; off >>= 1) {
        if (t < off) sred[t] += sred[t + off];
        __syncthreads();
    }
    float inv = 1.0f / sred[0];
    #pragma unroll
    for (int q = 0; q < 4; ++q) {
        int s = t + q * 256;
        gus[p * HW + s] = f2bf(g[q] * inv);
    }
}

// ---------- 2) SE mean: y[c] = mean_s x[c,s] ----------
__global__ void se_mean_kernel(const float* __restrict__ x, float* __restrict__ y) {
    __shared__ float sred[256];
    const int c = blockIdx.x, t = threadIdx.x;
    float a = 0.f;
    #pragma unroll
    for (int q = 0; q < 4; ++q) a += x[c * HW + t + q * 256];
    sred[t] = a; __syncthreads();
    for (int off = 128; off > 0; off >>= 1) {
        if (t < off) sred[t] += sred[t + off];
        __syncthreads();
    }
    if (t == 0) y[c] = sred[0] * (1.0f / 1024.0f);
}

// ---------- 3) SE MLP: y2 = sigmoid(W2 relu(W1 y + b1) + b2) ----------
__global__ void se_mlp_kernel(const float* __restrict__ y,
                              const float* __restrict__ w1, const float* __restrict__ b1,
                              const float* __restrict__ w2, const float* __restrict__ b2,
                              float* __restrict__ y2) {
    __shared__ float ys[1024];
    __shared__ float y1s[64];
    const int t = threadIdx.x;
    #pragma unroll
    for (int q = 0; q < 4; ++q) ys[t + q * 256] = y[t + q * 256];
    __syncthreads();
    if (t < 64) {
        float a = b1[t];
        for (int j = 0; j < 1024; ++j) a += w1[t * 1024 + j] * ys[j];
        y1s[t] = fmaxf(a, 0.f);
    }
    __syncthreads();
    #pragma unroll
    for (int q = 0; q < 4; ++q) {
        int c = t * 4 + q;
        float a = b2[c];
        for (int j = 0; j < 64; ++j) a += w2[c * 64 + j] * y1s[j];
        y2[c] = sigmoidf(a);
    }
}

// ---------- 4) apply SE scale: xse (bf16, GEMM1 A) and xr = sigmoid(xse) (f32) ----------
__global__ void se_apply_kernel(const float* __restrict__ x, const float* __restrict__ y2,
                                unsigned short* __restrict__ xse, float* __restrict__ xr) {
    const int c = blockIdx.x, t = threadIdx.x;
    const float sc = y2[c];
    #pragma unroll
    for (int q = 0; q < 4; ++q) {
        int s = t + q * 256;
        float v = x[c * HW + s] * sc;
        xse[c * HW + s] = f2bf(v);
        xr[c * HW + s] = sigmoidf(v);
    }
}

// ---------- 5) conv_w f32 -> bf16 ----------
__global__ void cvt_kernel(const float* __restrict__ src, unsigned short* __restrict__ dst, int n) {
    int i = blockIdx.x * blockDim.x + threadIdx.x;
    for (; i < n; i += gridDim.x * blockDim.x) dst[i] = f2bf(src[i]);
}

// ---------- 6) scrambled 3x3 softmax attention -> catT[p, 0..1023] ----------
__global__ void attn_kernel(const float* __restrict__ xr, unsigned short* __restrict__ catT) {
    __shared__ float red[9 * 256];
    __shared__ float attn[9];
    const int p = blockIdx.x;
    const int ph = p >> 5, pw = p & 31;
    const int t = threadIdx.x;

    float acc[9];
    #pragma unroll
    for (int ab = 0; ab < 9; ++ab) acc[ab] = 0.f;

    for (int q = 0; q < 4; ++q) {
        int d = t + q * 256;
        float x1v = xr[d * HW + p];
        #pragma unroll
        for (int ab = 0; ab < 9; ++ab) {
            int a = ab / 3, b = ab % 3;
            int f = a * 3072 + b * 1024 + d;      // scrambled reshape(hw,3,3,c)
            int ci = f / 9;
            int rem = f - ci * 9;
            int hh = ph + rem / 3 - 1;
            int ww = pw + rem % 3 - 1;
            float v = 0.f;
            if (hh >= 0 && hh < 32 && ww >= 0 && ww < 32) v = xr[ci * HW + hh * 32 + ww];
            acc[ab] += x1v * v;
        }
    }
    #pragma unroll
    for (int ab = 0; ab < 9; ++ab) red[ab * 256 + t] = acc[ab];
    __syncthreads();
    for (int off = 128; off > 0; off >>= 1) {
        if (t < off)
            for (int ab = 0; ab < 9; ++ab) red[ab * 256 + t] += red[ab * 256 + t + off];
        __syncthreads();
    }
    if (t == 0) {
        float mx = red[0];
        for (int ab = 1; ab < 9; ++ab) mx = fmaxf(mx, red[ab * 256]);
        float s = 0.f, e[9];
        for (int ab = 0; ab < 9; ++ab) { e[ab] = __expf(red[ab * 256] - mx); s += e[ab]; }
        float inv = 1.0f / s;
        for (int ab = 0; ab < 9; ++ab) attn[ab] = e[ab] * inv;
    }
    __syncthreads();

    for (int q = 0; q < 4; ++q) {
        int d = t + q * 256;
        float o = 0.f;
        #pragma unroll
        for (int ab = 0; ab < 9; ++ab) {
            int a = ab / 3, b = ab % 3;
            int f = a * 3072 + b * 1024 + d;
            int ci = f / 9;
            int rem = f - ci * 9;
            int hh = ph + rem / 3 - 1;
            int ww = pw + rem % 3 - 1;
            float v = 0.f;
            if (hh >= 0 && hh < 32 && ww >= 0 && ww < 32) v = xr[ci * HW + hh * 32 + ww];
            o += attn[ab] * v;
        }
        catT[p * K2 + d] = f2bf(o);
    }
}

// ---------- WMMA core: one wave computes a 32x64 strip of C = A[M,K] x BT[N,K]^T ----------
// Software-pipelined (register double-buffered) with pointer-bump addressing:
// all loads in an iteration use literal byte offsets; six base pointers advance
// by 64 elements (128 B) per iteration.
struct Fr { union { uint4 u[2]; v16bf v; }; };

template<int K>
__device__ __forceinline__ void gemm_strip32x64(const unsigned short* __restrict__ A,
                                                const unsigned short* __restrict__ BT,
                                                int m0, int n0, int lane, v8f acc[2][4]) {
    const int kbA = (lane >> 4) * 16;   // A: e0..15 -> K=kbA+e
    const int kbB = (lane >> 4) * 16;   // B: e0..15 -> K=kbB+e
    const int nc  = lane & 15;
    const unsigned short* ap0 = A + (size_t)(m0 + (lane & 15)) * K + kbA;
    const unsigned short* ap1 = ap0 + (size_t)16 * K;
    const unsigned short* bp0 = BT + (size_t)(n0 +  0 + nc) * K + kbB;
    const unsigned short* bp1 = BT + (size_t)(n0 + 16 + nc) * K + kbB;
    const unsigned short* bp2 = BT + (size_t)(n0 + 32 + nc) * K + kbB;
    const unsigned short* bp3 = BT + (size_t)(n0 + 48 + nc) * K + kbB;

    Fr a0[2], b0[4], a1[2], b1[4];

    auto loadA = [&](Fr* a, int off) {   // off in elements (compile-time constant)
        a[0].u[0] = *reinterpret_cast<const uint4*>(ap0 + off);
        a[0].u[1] = *reinterpret_cast<const uint4*>(ap0 + off + 8);
        a[1].u[0] = *reinterpret_cast<const uint4*>(ap1 + off);
        a[1].u[1] = *reinterpret_cast<const uint4*>(ap1 + off + 8);
    };
    auto loadB = [&](Fr* b, int off) {
        b[0].u[0] = *reinterpret_cast<const uint4*>(bp0 + off);
        b[0].u[1] = *reinterpret_cast<const uint4*>(bp0 + off + 8);
        b[1].u[0] = *reinterpret_cast<const uint4*>(bp1 + off);
        b[1].u[1] = *reinterpret_cast<const uint4*>(bp1 + off + 8);
        b[2].u[0] = *reinterpret_cast<const uint4*>(bp2 + off);
        b[2].u[1] = *reinterpret_cast<const uint4*>(bp2 + off + 8);
        b[3].u[0] = *reinterpret_cast<const uint4*>(bp3 + off);
        b[3].u[1] = *reinterpret_cast<const uint4*>(bp3 + off + 8);
    };
    auto mmas = [&](Fr* a, Fr* b) {
        #pragma unroll
        for (int mt = 0; mt < 2; ++mt)
            #pragma unroll
            for (int nt = 0; nt < 4; ++nt)
                acc[mt][nt] = __builtin_amdgcn_wmma_f32_16x16x32_bf16(
                    false, a[mt].v, false, b[nt].v, (short)0, acc[mt][nt], false, false);
    };

    loadA(a0, 0); loadB(b0, 0);
    #pragma unroll 1
    for (int k = 0; k < K; k += 64) {       // K is a multiple of 64
        loadA(a1, 32); loadB(b1, 32);
        mmas(a0, b0);
        if (k + 64 < K) { loadA(a0, 64); loadB(b0, 64); }
        mmas(a1, b1);
        ap0 += 64; ap1 += 64;
        bp0 += 64; bp1 += 64; bp2 += 64; bp3 += 64;
    }
}

// ---------- 7) GEMM1: x_spatial[c,p] = xse[c,:] . gus[p,:]  ->  catT[p, 1024+c] (bf16) ----------
__global__ void __launch_bounds__(128, 1)
gemm1_kernel(const unsigned short* __restrict__ A,   // xse [1024,1024]
             const unsigned short* __restrict__ BT,  // gus [1024,1024] = B^T
             unsigned short* __restrict__ catT) {
    const int wave = threadIdx.x >> 5, lane = threadIdx.x & 31;
    const int m0 = blockIdx.x * 128 + wave * 32;
    const int n0 = blockIdx.y * 64;
    v8f acc[2][4];
    const v8f z = {0.f,0.f,0.f,0.f,0.f,0.f,0.f,0.f};
    #pragma unroll
    for (int mt = 0; mt < 2; ++mt)
        #pragma unroll
        for (int nt = 0; nt < 4; ++nt) acc[mt][nt] = z;
    gemm_strip32x64<C_CH>(A, BT, m0, n0, lane, acc);
    #pragma unroll
    for (int mt = 0; mt < 2; ++mt)
        #pragma unroll
        for (int nt = 0; nt < 4; ++nt)
            #pragma unroll
            for (int r = 0; r < 8; ++r) {
                int m = m0 + mt * 16 + r + ((lane >> 4) << 3);  // channel c
                int n = n0 + nt * 16 + (lane & 15);             // pixel p
                catT[(size_t)n * K2 + C_CH + m] = f2bf(acc[mt][nt][r]);
            }
}

// ---------- 8) GEMM2: out = relu(BN(conv_w . cat)) ----------
__global__ void __launch_bounds__(128, 1)
gemm2_kernel(const unsigned short* __restrict__ A,   // conv_w bf16 [1024,2048]
             const unsigned short* __restrict__ BT,  // catT [1024,2048]
             const float* __restrict__ gamma, const float* __restrict__ beta,
             const float* __restrict__ mean,  const float* __restrict__ var,
             float* __restrict__ out) {
    const int wave = threadIdx.x >> 5, lane = threadIdx.x & 31;
    const int m0 = blockIdx.x * 128 + wave * 32;
    const int n0 = blockIdx.y * 64;
    v8f acc[2][4];
    const v8f z = {0.f,0.f,0.f,0.f,0.f,0.f,0.f,0.f};
    #pragma unroll
    for (int mt = 0; mt < 2; ++mt)
        #pragma unroll
        for (int nt = 0; nt < 4; ++nt) acc[mt][nt] = z;
    gemm_strip32x64<K2>(A, BT, m0, n0, lane, acc);
    #pragma unroll
    for (int mt = 0; mt < 2; ++mt)
        #pragma unroll
        for (int nt = 0; nt < 4; ++nt)
            #pragma unroll
            for (int r = 0; r < 8; ++r) {
                int m = m0 + mt * 16 + r + ((lane >> 4) << 3);  // out channel o
                int n = n0 + nt * 16 + (lane & 15);             // pixel p
                float sc = gamma[m] * rsqrtf(var[m] + 1e-5f);
                float v = (acc[mt][nt][r] - mean[m]) * sc + beta[m];
                out[(size_t)m * HW + n] = fmaxf(v, 0.f);
            }
}

extern "C" void kernel_launch(void* const* d_in, const int* in_sizes, int n_in,
                              void* d_out, int out_size, void* d_ws, size_t ws_size,
                              hipStream_t stream) {
    const float* x     = (const float*)d_in[0];
    const float* se_w1 = (const float*)d_in[1];
    const float* se_b1 = (const float*)d_in[2];
    const float* se_w2 = (const float*)d_in[3];
    const float* se_b2 = (const float*)d_in[4];
    const float* convw = (const float*)d_in[5];
    const float* bn_g  = (const float*)d_in[6];
    const float* bn_b  = (const float*)d_in[7];
    const float* bn_m  = (const float*)d_in[8];
    const float* bn_v  = (const float*)d_in[9];
    float* out = (float*)d_out;

    char* ws = (char*)d_ws;
    unsigned short* gus   = (unsigned short*)(ws);                          // 2 MB  [p,s]
    unsigned short* xse   = (unsigned short*)(ws + (2u << 20));             // 2 MB  [c,s]
    unsigned short* convb = (unsigned short*)(ws + (4u << 20));             // 4 MB  [o,2048]
    unsigned short* catT  = (unsigned short*)(ws + (8u << 20));             // 4 MB  [p,2048]
    float* xr             = (float*)(ws + (12u << 20));                     // 4 MB  [c,s]
    float* yv             = (float*)(ws + (16u << 20));                     // 4 KB
    float* y2             = (float*)(ws + (16u << 20) + 8192);              // 4 KB

    gauss_kernel   <<<1024, 256, 0, stream>>>(gus);
    se_mean_kernel <<<1024, 256, 0, stream>>>(x, yv);
    se_mlp_kernel  <<<1,    256, 0, stream>>>(yv, se_w1, se_b1, se_w2, se_b2, y2);
    se_apply_kernel<<<1024, 256, 0, stream>>>(x, y2, xse, xr);
    cvt_kernel     <<<2048, 256, 0, stream>>>(convw, convb, 1024 * 2048);
    attn_kernel    <<<1024, 256, 0, stream>>>(xr, catT);
    gemm1_kernel   <<<dim3(8, 16), 128, 0, stream>>>(xse, gus, catT);
    gemm2_kernel   <<<dim3(8, 16), 128, 0, stream>>>(convb, catT, bn_g, bn_b, bn_m, bn_v, out);
}
